// SplitNegativePositive_51101520888168
// MI455X (gfx1250) — matrix operand order repaired and checked by reading, not compiled, not run on hardware
//
#include <hip/hip_runtime.h>

// ---------------------------------------------------------------------------
// SplitNegativePositive on MI455X (gfx1250, wave32, WMMA)
//
// u_out = inputs * u_count, l_out = inputs * l_mask, masks from cosine
// thresholding against 512 prototype rows. Scores via v_wmma_f32_16x16x32_f16
// (f16 normalized operands, f32 accumulate); prototypes LDS-resident
// (XOR-swizzled); per-row hit counts via ballot_w32 + scalar popcount (SALU
// pipe, concurrent with WMMA); streaming outputs use non-temporal stores.
// ---------------------------------------------------------------------------

typedef _Float16 v4h  __attribute__((ext_vector_type(4)));
typedef _Float16 v8h  __attribute__((ext_vector_type(8)));
typedef _Float16 v16h __attribute__((ext_vector_type(16)));
typedef float    v8f  __attribute__((ext_vector_type(8)));
typedef float    v4f  __attribute__((ext_vector_type(4)));

#define D        128      // feature dim (halfs per row)
#define PROTOS   512      // 256 L + 256 U
#define THRESH   0.2f
#define EPSN     1e-12f

// --------------------------- kernel 1: normalize prototypes -> f16 ----------
__global__ __launch_bounds__(256) void proto_norm_kernel(
    const float* __restrict__ pl, const float* __restrict__ pu,
    _Float16* __restrict__ out, int Pl, int Ptot)
{
  int wave = (blockIdx.x * blockDim.x + threadIdx.x) >> 5;   // one wave per row
  int lane = threadIdx.x & 31;
  if (wave >= Ptot) return;
  const float* src = (wave < Pl) ? (pl + (size_t)wave * D)
                                 : (pu + (size_t)(wave - Pl) * D);
  float4 x = ((const float4*)src)[lane];                     // 4 f32 per lane
  float ss = x.x * x.x + x.y * x.y + x.z * x.z + x.w * x.w;
#pragma unroll
  for (int off = 16; off > 0; off >>= 1) ss += __shfl_xor(ss, off, 32);
  float inv = rsqrtf(fmaxf(ss, EPSN));
  v4h h = { (_Float16)(x.x * inv), (_Float16)(x.y * inv),
            (_Float16)(x.z * inv), (_Float16)(x.w * inv) };
  *(v4h*)(out + (size_t)wave * D + lane * 4) = h;
}

// --------------------------- kernel 2: main ---------------------------------
// LDS layout (dynamic, 163840 B):
//   s_proto: 512 rows x 128 halfs, 16B chunks XOR-swizzled by (row & 15)
//   s_a    : 8 waves x 16 rows x 128 halfs, same swizzle (key = row-in-tile)
__global__ __launch_bounds__(256) void cosine_split_kernel(
    const float* __restrict__ inp, const _Float16* __restrict__ protoh,
    float* __restrict__ uout, float* __restrict__ lout,
    int Nrows, int numTiles, int Ptot)
{
  extern __shared__ _Float16 smem[];
  _Float16* s_proto = smem;                 // PROTOS * D halfs
  _Float16* s_a     = smem + PROTOS * D;    // 8 * 16 * D halfs

  const int tid  = threadIdx.x;
  const int lane = tid & 31;
  const int wave = tid >> 5;
  const int h    = lane >> 4;               // half-wave id (0/1)
  const int nl   = lane & 15;

  // ---- stage all prototypes into LDS (16B chunks, XOR swizzle) ----
  int nChunks = Ptot * (D / 8);             // 8 halfs = 16 B per chunk
  for (int c = tid; c < nChunks; c += 256) {
    int row = c >> 4, ch = c & 15;
    uint4 v = ((const uint4*)protoh)[c];
    ((uint4*)s_proto)[row * 16 + (ch ^ (row & 15))] = v;
  }
  __syncthreads();

  _Float16* s_aw = s_a + wave * (16 * D);
  const int totalWaves = gridDim.x * 8;

  for (int tile = blockIdx.x * 8 + wave; tile < numTiles; tile += totalWaves) {
    const int row0 = tile * 16;

    // ---- normalize 16 input rows -> f16 A panel in LDS ----
    for (int m = 0; m < 16; ++m) {
      int r = row0 + m;
      float4 x = make_float4(0.f, 0.f, 0.f, 0.f);
      if (r < Nrows) x = ((const float4*)inp)[(size_t)r * 32 + lane];
      float ss = x.x * x.x + x.y * x.y + x.z * x.z + x.w * x.w;
#pragma unroll
      for (int off = 16; off > 0; off >>= 1) ss += __shfl_xor(ss, off, 32);
      float inv = rsqrtf(fmaxf(ss, EPSN));
      v4h hx = { (_Float16)(x.x * inv), (_Float16)(x.y * inv),
                 (_Float16)(x.z * inv), (_Float16)(x.w * inv) };
      int dst = m * D + (((lane >> 1) ^ m) * 8) + (lane & 1) * 4;
      *(v4h*)(s_aw + dst) = hx;
    }
    // same-wave DS RAW: DS ops are in-order per wave; block compiler reordering
    asm volatile("s_wait_dscnt 0" ::: "memory");
    __builtin_amdgcn_wave_barrier();

    // ---- load A fragments (16x32 f16 per K-chunk; contiguous k order, B matches) ----
    // lane (m = lane&15, khalf = h*16) holds k = khalf..khalf+15
    v16h a[4];
#pragma unroll
    for (int c = 0; c < 4; ++c) {
      int ch0 = (4 * c + 2 * h)     ^ nl;
      int ch1 = (4 * c + 2 * h + 1) ^ nl;
      v8h lo = *(const v8h*)(s_aw + nl * D + ch0 * 8);
      v8h hi = *(const v8h*)(s_aw + nl * D + ch1 * 8);
      a[c] = __builtin_shufflevector(lo, hi, 0,1,2,3,4,5,6,7,8,9,10,11,12,13,14,15);
    }

    // Uniform (SGPR) hit counters: Lo -> rows m=0..7, Hi -> rows m=8..15.
    int lcLo[8] = {0,0,0,0,0,0,0,0}, lcHi[8] = {0,0,0,0,0,0,0,0};
    int ucLo[8] = {0,0,0,0,0,0,0,0}, ucHi[8] = {0,0,0,0,0,0,0,0};

    // ---- prototype tiles in pairs: two independent WMMA accumulator chains.
    // loss = -dot >= THRESH  <=>  dot <= -THRESH.
    // C/D layout: lane bits 0-15 hold N=0..15 of rows m=v; bits 16-31 rows m=v+8,
    // so ballot_w32 of the compare gives both per-row 16-wide hit masks at once.
#pragma unroll 1
    for (int t = 0; t < 16; t += 2) {           // L set: prototypes 0..255
      v8f acc0 = {}, acc1 = {};
#pragma unroll
      for (int c = 0; c < 4; ++c) {
        int pr0 = t * 16 + nl, pr1 = pr0 + 16;
        int ch0 = (4 * c + 2 * h) ^ nl, ch1 = ch0 ^ 1;
        v8h lo0 = *(const v8h*)(s_proto + pr0 * D + ch0 * 8);
        v8h hi0 = *(const v8h*)(s_proto + pr0 * D + ch1 * 8);
        v8h lo1 = *(const v8h*)(s_proto + pr1 * D + ch0 * 8);
        v8h hi1 = *(const v8h*)(s_proto + pr1 * D + ch1 * 8);
        v16h b0 = __builtin_shufflevector(lo0, hi0, 0,1,2,3,4,5,6,7,8,9,10,11,12,13,14,15);
        v16h b1 = __builtin_shufflevector(lo1, hi1, 0,1,2,3,4,5,6,7,8,9,10,11,12,13,14,15);
        acc0 = __builtin_amdgcn_wmma_f32_16x16x32_f16(false, a[c], false, b0, (short)0, acc0, false, false);
        acc1 = __builtin_amdgcn_wmma_f32_16x16x32_f16(false, a[c], false, b1, (short)0, acc1, false, false);
      }
#pragma unroll
      for (int v = 0; v < 8; ++v) {
        unsigned m0 = __builtin_amdgcn_ballot_w32(acc0[v] <= -THRESH);
        unsigned m1 = __builtin_amdgcn_ballot_w32(acc1[v] <= -THRESH);
        lcLo[v] += __builtin_popcount(m0 & 0xffffu) + __builtin_popcount(m1 & 0xffffu);
        lcHi[v] += __builtin_popcount(m0 >> 16)     + __builtin_popcount(m1 >> 16);
      }
    }
#pragma unroll 1
    for (int t = 16; t < 32; t += 2) {          // U set: prototypes 256..511
      v8f acc0 = {}, acc1 = {};
#pragma unroll
      for (int c = 0; c < 4; ++c) {
        int pr0 = t * 16 + nl, pr1 = pr0 + 16;
        int ch0 = (4 * c + 2 * h) ^ nl, ch1 = ch0 ^ 1;
        v8h lo0 = *(const v8h*)(s_proto + pr0 * D + ch0 * 8);
        v8h hi0 = *(const v8h*)(s_proto + pr0 * D + ch1 * 8);
        v8h lo1 = *(const v8h*)(s_proto + pr1 * D + ch0 * 8);
        v8h hi1 = *(const v8h*)(s_proto + pr1 * D + ch1 * 8);
        v16h b0 = __builtin_shufflevector(lo0, hi0, 0,1,2,3,4,5,6,7,8,9,10,11,12,13,14,15);
        v16h b1 = __builtin_shufflevector(lo1, hi1, 0,1,2,3,4,5,6,7,8,9,10,11,12,13,14,15);
        acc0 = __builtin_amdgcn_wmma_f32_16x16x32_f16(false, a[c], false, b0, (short)0, acc0, false, false);
        acc1 = __builtin_amdgcn_wmma_f32_16x16x32_f16(false, a[c], false, b1, (short)0, acc1, false, false);
      }
#pragma unroll
      for (int v = 0; v < 8; ++v) {
        unsigned m0 = __builtin_amdgcn_ballot_w32(acc0[v] <= -THRESH);
        unsigned m1 = __builtin_amdgcn_ballot_w32(acc1[v] <= -THRESH);
        ucLo[v] += __builtin_popcount(m0 & 0xffffu) + __builtin_popcount(m1 & 0xffffu);
        ucHi[v] += __builtin_popcount(m0 >> 16)     + __builtin_popcount(m1 >> 16);
      }
    }

    // ---- stream outputs: scales are wave-uniform scalars; NT stores so the
    // 205 MB write stream doesn't evict prototypes/inputs from WGP$/L2.
#pragma unroll
    for (int m = 0; m < 16; ++m) {
      int lc = (m < 8) ? lcLo[m] : lcHi[m - 8];
      int uc = (m < 8) ? ucLo[m] : ucHi[m - 8];
      float ls = (lc > 0) ? 1.f : 0.f;
      float us = (lc > 0) ? 0.f : (float)uc;
      int r = row0 + m;
      if (r < Nrows) {
        const v4f* xin = (const v4f*)inp + ((size_t)r * 32 + lane);
        v4f x = __builtin_nontemporal_load(xin);   // last-use re-read
        v4f u = { x[0] * us, x[1] * us, x[2] * us, x[3] * us };
        v4f l = { x[0] * ls, x[1] * ls, x[2] * ls, x[3] * ls };
        __builtin_nontemporal_store(u, (v4f*)uout + ((size_t)r * 32 + lane));
        __builtin_nontemporal_store(l, (v4f*)lout + ((size_t)r * 32 + lane));
      }
    }
  }
}

// --------------------------- launcher ---------------------------------------
extern "C" void kernel_launch(void* const* d_in, const int* in_sizes, int n_in,
                              void* d_out, int out_size, void* d_ws, size_t ws_size,
                              hipStream_t stream) {
  const float* pl  = (const float*)d_in[0];
  const float* pu  = (const float*)d_in[1];
  const float* inp = (const float*)d_in[2];
  int Pl    = in_sizes[0] / D;       // 256
  int Pu    = in_sizes[1] / D;       // 256
  int Ptot  = Pl + Pu;               // 512
  int Nrows = in_sizes[2] / D;       // 200000

  float* uout = (float*)d_out;
  float* lout = uout + (size_t)Nrows * D;
  _Float16* protoh = (_Float16*)d_ws;  // 512*128 f16 = 128 KB

  int blocks1 = (Ptot + 7) / 8;        // 8 waves/block, 1 wave per row
  proto_norm_kernel<<<blocks1, 256, 0, stream>>>(pl, pu, protoh, Pl, Ptot);

  int numTiles = (Nrows + 15) / 16;    // 12500
  int grid = 512;
  if (grid * 8 > numTiles) grid = (numTiles + 7) / 8;
  size_t shmem = (size_t)(PROTOS * D + 8 * 16 * D) * sizeof(_Float16); // 160 KB
  cosine_split_kernel<<<grid, 256, shmem, stream>>>(
      inp, protoh, uout, lout, Nrows, numTiles, Ptot);
}
